// ModelMamba_38354057953799
// MI455X (gfx1250) — compile-verified, matrix operand order checked by reading, not run on hardware
//
#include <hip/hip_runtime.h>

// ---------------- problem constants ----------------
constexpr int Bsz    = 16;
constexpr int Lseq   = 1024;
constexpr int D_TIS  = 64;
constexpr int D_TOK  = 192;
constexpr int D_MODEL= 256;   // D_TIS + D_TOK
constexpr int D_STATE= 16;
constexpr int D_CONV = 4;
constexpr int D_INNER= 512;
constexpr int DT_RANK= 16;
constexpr int XDBL   = DT_RANK + 2*D_STATE;  // 48
constexpr int HIDDEN = 512;
constexpr int BL     = Bsz * Lseq;           // 16384

// ---------------- WMMA types ----------------
typedef __attribute__((ext_vector_type(16))) __bf16 v16bf;
typedef __attribute__((ext_vector_type(8)))  __bf16 v8bf;
typedef __attribute__((ext_vector_type(8)))  float  v8f;

// ---------------- helpers ----------------
__global__ __launch_bounds__(256) void f2bf_kernel(const float* __restrict__ src,
                                                   __bf16* __restrict__ dst, int n) {
    int i = blockIdx.x * blockDim.x + threadIdx.x;
    if (i < n) dst[i] = (__bf16)src[i];
}

// embedding concat: u[b,l, 0:192]=seq_emb[tok], u[b,l,192:256]=tissue_emb[tid]
__global__ __launch_bounds__(256)
void embed_kernel(const int* __restrict__ rna, const int* __restrict__ tissue_id,
                  const float* __restrict__ tissue_emb, const float* __restrict__ seq_emb,
                  __bf16* __restrict__ u_bf) {
    int bl = blockIdx.x;            // (b,l)
    int b  = bl >> 10;
    int c  = threadIdx.x;           // channel 0..255
    float v;
    if (c < D_TOK) {
        int tok = rna[bl];
        v = seq_emb[tok * D_TOK + c];
    } else {
        int t = tissue_id[b];
        v = tissue_emb[t * D_TIS + (c - D_TOK)];
    }
    u_bf[(size_t)bl * D_MODEL + c] = (__bf16)v;
}

// ---------------- WMMA bf16 fragment loads ----------------
// ISA 7.12.2 16-bit operand layout (A 16x32, and W rows fed identically for B):
// lane: m = lane&15, lg = lane>>4; v16bf element e holds K = 16*(e>=8) + 8*lg + (e&7).
// => two contiguous 16-byte chunks per lane: [kb + 8*lg, +8) and [kb + 16 + 8*lg, +8)
__device__ __forceinline__
v16bf load_frag(const __bf16* __restrict__ base, int ld, int lane, int kb) {
    const int mr = lane & 15;
    const int lg = lane >> 4;
    const __bf16* p = base + (size_t)mr * ld + kb + 8 * lg;
    v8bf lo = *(const v8bf*)p;          // global_load_b128
    v8bf hi = *(const v8bf*)(p + 16);   // global_load_b128
    return __builtin_shufflevector(lo, hi, 0,1,2,3,4,5,6,7,8,9,10,11,12,13,14,15);
}

// K==16 special case: elements 0..7 are K=8*lg+i (always < 16, in-bounds),
// elements 8..15 are K>=16 (always zero-pad). No conditional loads needed.
__device__ __forceinline__
v16bf load_frag_k16(const __bf16* __restrict__ base, int ld, int lane) {
    const int mr = lane & 15;
    const int lg = lane >> 4;
    v8bf lo = *(const v8bf*)(base + (size_t)mr * ld + 8 * lg);  // global_load_b128
    v8bf zz = {};
    return __builtin_shufflevector(lo, zz, 0,1,2,3,4,5,6,7,8,9,10,11,12,13,14,15);
}

// ---------------- WMMA bf16 GEMM:  C[M,N] = A[M,K] * W[N,K]^T ----------------
// K is compile-time (fully unrolled, branch-free). MT = M tiles per wave (B reuse).
// MODE 0: plain   MODE 1: softplus(c + bias[n])   MODE 2: relu(c + bias[n])
template <int MODE, int K, int MT>
__global__ __launch_bounds__(256)
void gemm_bf16_wmma(const __bf16* __restrict__ A, const __bf16* __restrict__ W,
                    float* __restrict__ C, __bf16* __restrict__ Cbf,
                    const float* __restrict__ bias,
                    int N, int lda, int ldw, int ldc) {
    const int lane = threadIdx.x & 31;
    const int wave = threadIdx.x >> 5;
    const int m0 = blockIdx.y * (16 * MT);
    const int n0 = blockIdx.x * 128 + wave * 16;
    if (n0 >= N) return;   // wave-uniform exit (EXEC stays all-1 for WMMA)

    v8f acc[MT] = {};
    const __bf16* Wb = W + (size_t)n0 * ldw;

    if constexpr (K % 32 == 0) {
#pragma unroll
        for (int kb = 0; kb < K; kb += 32) {
            v16bf b = load_frag(Wb, ldw, lane, kb);
#pragma unroll
            for (int mt = 0; mt < MT; ++mt) {
                v16bf a = load_frag(A + (size_t)(m0 + 16 * mt) * lda, lda, lane, kb);
                acc[mt] = __builtin_amdgcn_wmma_f32_16x16x32_bf16(
                              false, a, false, b, (short)0, acc[mt], false, false);
            }
        }
    } else {  // K == 16, zero-padded upper half, single WMMA per tile
        v16bf b = load_frag_k16(Wb, ldw, lane);
#pragma unroll
        for (int mt = 0; mt < MT; ++mt) {
            v16bf a = load_frag_k16(A + (size_t)(m0 + 16 * mt) * lda, lda, lane);
            acc[mt] = __builtin_amdgcn_wmma_f32_16x16x32_bf16(
                          false, a, false, b, (short)0, acc[mt], false, false);
        }
    }

    // C layout: VGPR r -> M = mtile + r + 8*(lane>>4), N = n0 + (lane&15)
    const int n  = n0 + (lane & 15);
    const int mh = 8 * (lane >> 4);
#pragma unroll
    for (int mt = 0; mt < MT; ++mt) {
        const int mb = m0 + 16 * mt + mh;
#pragma unroll
        for (int r = 0; r < 8; ++r) {
            float v = acc[mt][r];
            if (MODE == 1) { v += bias[n]; v = (v > 20.f) ? v : log1pf(expf(v)); }
            if (MODE == 2) { v += bias[n]; v = v > 0.f ? v : 0.f; }
            size_t idx = (size_t)(mb + r) * ldc + n;
            C[idx] = v;
            if (Cbf) Cbf[idx] = (__bf16)v;
        }
    }
}

// ---------------- causal depthwise conv(4) + bias + SiLU ----------------
__global__ __launch_bounds__(512)
void conv_silu_kernel(const float* __restrict__ xz, const float* __restrict__ conv_w,
                      const float* __restrict__ conv_b,
                      float* __restrict__ xc, __bf16* __restrict__ xc_bf) {
    int bl = blockIdx.x;            // (b,l)
    int d  = threadIdx.x;           // channel 0..511
    int b  = bl >> 10;
    int l  = bl & (Lseq - 1);
    float acc = conv_b[d];
#pragma unroll
    for (int j = 0; j < D_CONV; ++j) {
        int lj = l - (D_CONV - 1) + j;
        if (lj >= 0)
            acc += conv_w[d * D_CONV + j] *
                   xz[((size_t)b * Lseq + lj) * (2 * D_INNER) + d];
    }
    float s = acc / (1.f + expf(-acc));          // SiLU
    size_t idx = (size_t)bl * D_INNER + d;
    xc[idx]    = s;
    xc_bf[idx] = (__bf16)s;
}

// ---------------- serial selective scan, fused +x*D and SiLU(z) gate ----------------
__global__ __launch_bounds__(512)
void scan_kernel(const float* __restrict__ dt, const float* __restrict__ x_dbl,
                 const float* __restrict__ xc, const float* __restrict__ xz,
                 const float* __restrict__ A_log, const float* __restrict__ Dp,
                 __bf16* __restrict__ y_bf) {
    const int b = blockIdx.x;       // batch
    const int d = threadIdx.x;      // inner channel 0..511
    __shared__ float sBC[2 * D_STATE];   // B_t then C_t

    float A[D_STATE], h[D_STATE];
#pragma unroll
    for (int n = 0; n < D_STATE; ++n) {
        A[n] = -expf(A_log[d * D_STATE + n]);
        h[n] = 0.f;
    }
    const float Dd = Dp[d];

    for (int t = 0; t < Lseq; ++t) {
        const size_t bl = (size_t)b * Lseq + t;
        if (d < 2 * D_STATE) sBC[d] = x_dbl[bl * XDBL + DT_RANK + d];
        __syncthreads();
        const float dtv = dt[bl * D_INNER + d];
        const float xv  = xc[bl * D_INNER + d];
        const float dx  = dtv * xv;
        float y = 0.f;
#pragma unroll
        for (int n = 0; n < D_STATE; ++n) {
            float dA = expf(dtv * A[n]);
            h[n] = dA * h[n] + dx * sBC[n];
            y += h[n] * sBC[D_STATE + n];
        }
        y += xv * Dd;
        float z = xz[bl * (2 * D_INNER) + D_INNER + d];
        y *= z / (1.f + expf(-z));               // * SiLU(z)
        y_bf[bl * D_INNER + d] = (__bf16)y;
        __syncthreads();
    }
}

// gather out[b, seq_len[b]-1, :] -> bf16 row for the head GEMM
__global__ __launch_bounds__(256)
void gather_kernel(const float* __restrict__ out, const int* __restrict__ seq_lengths,
                   __bf16* __restrict__ out_last_bf) {
    int b = blockIdx.x, c = threadIdx.x;
    int l = seq_lengths[b] - 1;
    out_last_bf[b * D_MODEL + c] =
        (__bf16)out[((size_t)b * Lseq + l) * D_MODEL + c];
}

// final: out[b] = hid[b,:] . w2 + b2
__global__ __launch_bounds__(256)
void head2_kernel(const float* __restrict__ hid, const float* __restrict__ w2,
                  const float* __restrict__ b2, float* __restrict__ outp) {
    __shared__ float red[256];
    int b = blockIdx.x, t = threadIdx.x;
    float acc = 0.f;
    for (int k = t; k < HIDDEN; k += 256) acc += hid[b * HIDDEN + k] * w2[k];
    red[t] = acc; __syncthreads();
    for (int s = 128; s > 0; s >>= 1) { if (t < s) red[t] += red[t + s]; __syncthreads(); }
    if (t == 0) outp[b] = red[0] + b2[0];
}

// ---------------- host launcher ----------------
extern "C" void kernel_launch(void* const* d_in, const int* in_sizes, int n_in,
                              void* d_out, int out_size, void* d_ws, size_t ws_size,
                              hipStream_t stream) {
    const int*   rna        = (const int*)  d_in[0];
    const int*   tissue_id  = (const int*)  d_in[1];
    const int*   seq_len    = (const int*)  d_in[2];
    const float* tissue_emb = (const float*)d_in[3];
    const float* seq_emb    = (const float*)d_in[4];
    const float* w_in       = (const float*)d_in[5];
    const float* conv_w     = (const float*)d_in[6];
    const float* conv_b     = (const float*)d_in[7];
    const float* w_x        = (const float*)d_in[8];
    const float* w_dt       = (const float*)d_in[9];
    const float* b_dt       = (const float*)d_in[10];
    const float* A_log      = (const float*)d_in[11];
    const float* Dp         = (const float*)d_in[12];
    const float* w_out      = (const float*)d_in[13];
    const float* w1         = (const float*)d_in[14];
    const float* b1         = (const float*)d_in[15];
    const float* w2         = (const float*)d_in[16];
    const float* b2         = (const float*)d_in[17];
    float* outp = (float*)d_out;

    char* ws = (char*)d_ws;
    size_t off = 0;
    auto alloc = [&](size_t bytes) -> void* {
        void* p = ws + off;
        off += (bytes + 255) & ~(size_t)255;
        return p;
    };

    __bf16* u_bf      = (__bf16*)alloc((size_t)BL * D_MODEL * 2);
    __bf16* w_in_bf   = (__bf16*)alloc((size_t)2 * D_INNER * D_MODEL * 2);
    __bf16* w_x_bf    = (__bf16*)alloc((size_t)XDBL * D_INNER * 2);
    __bf16* w_dt_bf   = (__bf16*)alloc((size_t)D_INNER * DT_RANK * 2);
    __bf16* w_out_bf  = (__bf16*)alloc((size_t)D_MODEL * D_INNER * 2);
    __bf16* w1_bf     = (__bf16*)alloc((size_t)HIDDEN * D_MODEL * 2);
    float*  xz        = (float*) alloc((size_t)BL * 2 * D_INNER * 4);
    float*  xc        = (float*) alloc((size_t)BL * D_INNER * 4);
    __bf16* xc_bf     = (__bf16*)alloc((size_t)BL * D_INNER * 2);
    float*  x_dbl     = (float*) alloc((size_t)BL * XDBL * 4);
    __bf16* x_dbl_bf  = (__bf16*)alloc((size_t)BL * XDBL * 2);
    float*  dt        = (float*) alloc((size_t)BL * D_INNER * 4);
    __bf16* y_bf      = (__bf16*)alloc((size_t)BL * D_INNER * 2);
    float*  out_f     = (float*) alloc((size_t)BL * D_MODEL * 4);
    __bf16* last_bf   = (__bf16*)alloc((size_t)Bsz * D_MODEL * 2);
    float*  hid       = (float*) alloc((size_t)Bsz * HIDDEN * 4);

    auto cvt = [&](const float* s, __bf16* d, int n) {
        f2bf_kernel<<<(n + 255) / 256, 256, 0, stream>>>(s, d, n);
    };
    cvt(w_in,  w_in_bf,  2 * D_INNER * D_MODEL);
    cvt(w_x,   w_x_bf,   XDBL * D_INNER);
    cvt(w_dt,  w_dt_bf,  D_INNER * DT_RANK);
    cvt(w_out, w_out_bf, D_MODEL * D_INNER);
    cvt(w1,    w1_bf,    HIDDEN * D_MODEL);

    // 1) embedding concat -> u_bf (BL x 256)
    embed_kernel<<<BL, 256, 0, stream>>>(rna, tissue_id, tissue_emb, seq_emb, u_bf);

    // 2) xz = u * w_in^T   (16384 x 1024, K=256), 32 rows per block
    gemm_bf16_wmma<0, D_MODEL, 2><<<dim3(8, BL / 32), 256, 0, stream>>>(
        u_bf, w_in_bf, xz, nullptr, nullptr,
        2 * D_INNER, D_MODEL, D_MODEL, 2 * D_INNER);

    // 3) causal depthwise conv + SiLU
    conv_silu_kernel<<<BL, D_INNER, 0, stream>>>(xz, conv_w, conv_b, xc, xc_bf);

    // 4) x_dbl = xc * w_x^T   (16384 x 48, K=512)
    gemm_bf16_wmma<0, D_INNER, 2><<<dim3(1, BL / 32), 256, 0, stream>>>(
        xc_bf, w_x_bf, x_dbl, x_dbl_bf, nullptr,
        XDBL, D_INNER, D_INNER, XDBL);

    // 5) dt = softplus(x_dbl[:, :16] * w_dt^T + b_dt)   (16384 x 512, K=16 zero-padded)
    gemm_bf16_wmma<1, DT_RANK, 2><<<dim3(4, BL / 32), 256, 0, stream>>>(
        x_dbl_bf, w_dt_bf, dt, nullptr, b_dt,
        D_INNER, XDBL, DT_RANK, D_INNER);

    // 6) serial scan (fused +x*D, *SiLU(z)) -> y_bf
    scan_kernel<<<Bsz, D_INNER, 0, stream>>>(dt, x_dbl, xc, xz, A_log, Dp, y_bf);

    // 7) out = y * w_out^T   (16384 x 256, K=512)
    gemm_bf16_wmma<0, D_INNER, 2><<<dim3(2, BL / 32), 256, 0, stream>>>(
        y_bf, w_out_bf, out_f, nullptr, nullptr,
        D_MODEL, D_INNER, D_INNER, D_MODEL);

    // 8) gather last valid timestep
    gather_kernel<<<Bsz, D_MODEL, 0, stream>>>(out_f, seq_len, last_bf);

    // 9) hid = relu(last * w1^T + b1)   (16 x 512, K=256), M=16 -> MT=1
    gemm_bf16_wmma<2, D_MODEL, 1><<<dim3(4, 1), 256, 0, stream>>>(
        last_bf, w1_bf, hid, nullptr, b1,
        HIDDEN, D_MODEL, D_MODEL, HIDDEN);

    // 10) final dot -> (16,1)
    head2_kernel<<<Bsz, 256, 0, stream>>>(hid, w2, b2, outp);
}
